// WorkingMemoryModel_29231547416771
// MI455X (gfx1250) — matrix operand order repaired and checked by reading, not compiled, not run on hardware
//
#include <hip/hip_runtime.h>
#include <hip/hip_bf16.h>

#define D   768
#define NS  256
#define ETA 1e-3f
#define DEC 0.99f

typedef __bf16 bf16;
typedef __attribute__((ext_vector_type(8)))  bf16  v8bf;
typedef __attribute__((ext_vector_type(16))) bf16  v16bf;
typedef __attribute__((ext_vector_type(8)))  float v8f;

static __device__ __forceinline__ bf16 f2bf(float f) {
  unsigned u = __builtin_bit_cast(unsigned, f);
  u += 0x7FFFu + ((u >> 16) & 1u);                 // round-to-nearest-even
  unsigned short h = (unsigned short)(u >> 16);
  return __builtin_bit_cast(bf16, h);
}
static __device__ __forceinline__ float bf2f(bf16 b) {
  unsigned u = ((unsigned)__builtin_bit_cast(unsigned short, b)) << 16;
  return __builtin_bit_cast(float, u);
}
static __device__ __forceinline__ v8bf load8(const bf16* p) {
  return *reinterpret_cast<const v8bf*>(p);
}
static __device__ __forceinline__ v16bf cat16(v8bf lo, v8bf hi) {
  return __builtin_shufflevector(lo, hi, 0,1,2,3,4,5,6,7,8,9,10,11,12,13,14,15);
}

// ---------------- elementwise helpers ----------------
__global__ void k_cvt(const float* __restrict__ src, bf16* __restrict__ dst, int n) {
  int i = blockIdx.x * blockDim.x + threadIdx.x;
  if (i < n) dst[i] = f2bf(src[i]);
}
// Wt[n*D + k] = W[k*D + n]  (so GEMM B-operand loads are contiguous per output col)
__global__ void k_transpose_cvt(const float* __restrict__ W, bf16* __restrict__ Wt) {
  int i = blockIdx.x * blockDim.x + threadIdx.x;
  if (i < D * D) { int n = i / D, k = i % D; Wt[i] = f2bf(W[k * D + n]); }
}
__global__ void k_zero_bf(bf16* __restrict__ p, int n) {
  int i = blockIdx.x * blockDim.x + threadIdx.x;
  if (i < n) p[i] = f2bf(0.f);
}

// ---------------- WMMA GEMM: C = act(A[MxK]_bf16 * B[KxN] + bias) ----------------
// Bt is B transposed, [N x K] row-major bf16 (contiguous K per output column).
// Each wave computes a 16x32 output (2 N-tiles sharing one A fragment); 8 waves/block.
__global__ __launch_bounds__(256) void k_gemm(
    const bf16* __restrict__ A, const bf16* __restrict__ Bt,
    const float* __restrict__ bias, float* __restrict__ outF,
    bf16* __restrict__ outBf, int M, int N, int K, int act)
{
  const int lane = threadIdx.x & 31;
  const int wave = threadIdx.x >> 5;
  const int r    = lane & 15;         // row (A) / col (B,C) within tile
  const int hi   = lane >> 4;         // K-half selector (contiguous k = 16*hi + s order used here)
  const int nblk = N >> 5;            // 32-wide N blocks
  const int task = blockIdx.x * 8 + wave;
  if (task >= (M >> 4) * nblk) return;
  const int tm = task / nblk;
  const int tn = (task % nblk) * 2;

  const bf16* Ar  = A  + (size_t)(tm * 16 + r) * K;            // A row, row-major
  const bf16* Bp0 = Bt + (size_t)(tn * 16 + r) * K + hi * 16;  // Bt row = out col
  const bf16* Bp1 = Bp0 + (size_t)16 * K;

  v8f acc0 = {0.f,0.f,0.f,0.f,0.f,0.f,0.f,0.f};
  v8f acc1 = {0.f,0.f,0.f,0.f,0.f,0.f,0.f,0.f};
  for (int kk = 0; kk < K; kk += 32) {
    __builtin_prefetch(Ar  + kk + 128, 0, 1);
    __builtin_prefetch(Bp0 + kk + 128, 0, 1);
    __builtin_prefetch(Bp1 + kk + 128, 0, 1);
    // 16-bit A 16x32 fragment: hi-half selects K 0..15 vs 16..31
    v16bf a  = cat16(load8(Ar + kk + hi * 16), load8(Ar + kk + hi * 16 + 8));
    // 16-bit B 32x16 fragment: lanes 0-15 hold K 0..15, lanes 16-31 hold K 16..31
    v16bf b0 = cat16(load8(Bp0 + kk), load8(Bp0 + kk + 8));
    v16bf b1 = cat16(load8(Bp1 + kk), load8(Bp1 + kk + 8));
    acc0 = __builtin_amdgcn_wmma_f32_16x16x32_bf16(false, a, false, b0,
                                                   (short)0, acc0, false, false);
    acc1 = __builtin_amdgcn_wmma_f32_16x16x32_bf16(false, a, false, b1,
                                                   (short)0, acc1, false, false);
  }
  const int col0 = tn * 16 + r;
  const int col1 = col0 + 16;
  const float bv0 = bias ? bias[col0] : 0.f;
  const float bv1 = bias ? bias[col1] : 0.f;
#pragma unroll
  for (int i = 0; i < 8; ++i) {                // C/D layout: VGPR i -> row i + 8*hi
    int row = tm * 16 + i + hi * 8;
    float v0 = acc0[i] + bv0;
    float v1 = acc1[i] + bv1;
    if (act == 1) { v0 = fmaxf(v0, 0.f); v1 = fmaxf(v1, 0.f); }
    if (outF)  { outF [(size_t)row * N + col0] = v0;
                 outF [(size_t)row * N + col1] = v1; }
    if (outBf) { outBf[(size_t)row * N + col0] = f2bf(v0);
                 outBf[(size_t)row * N + col1] = f2bf(v1); }
  }
}

// ---------------- chunked linear-attention scan ----------------
// H_t = act( Z_t + eta * sum_{s<t} DEC^(t-1-s) * G[t,s] * H_s )
// Cross-chunk history term via WMMA (Coef[16 x S] * H[S x D]); intra-chunk 16-step
// recurrence is per-column independent (registers, no barriers).
// G's 16-row panel for the chunk (16 KB, contiguous) is staged to LDS with
// GLOBAL_LOAD_ASYNC_TO_LDS_B128 (ASYNCcnt) and consumed twice from LDS.
// Single workgroup, 768 threads = 24 waves; 48 N-tiles -> 2 tiles per wave.
__global__ __launch_bounds__(768) void k_scan(
    const float* __restrict__ Z, const float* __restrict__ G,
    bf16* __restrict__ HT /* [D x NS] transposed history, zero-initialized */,
    bf16* __restrict__ outBf, float* __restrict__ outF, int act)
{
  __shared__ bf16  Pl[16][D];               // cross-chunk contribution (24 KB)
  __shared__ bf16  coef[16][NS];            // eta*DEC^(t-1-s)*G[t,s], zero-pad (8 KB)
  __shared__ alignas(16) float Gl[16][NS];  // chunk's G rows, async-staged (16 KB)
  __shared__ float dp[NS];                  // decay powers

  const int tid  = threadIdx.x;
  const int lane = tid & 31, wave = tid >> 5;
  const int r    = lane & 15, hi = lane >> 4;

  if (tid < NS) dp[tid] = __powf(DEC, (float)tid);
  __syncthreads();

  for (int c = 0; c < NS / 16; ++c) {
    const int S = c * 16;

    // --- async-stage G rows [S, S+16) (contiguous 16 KB) into LDS ---
    {
      float* GlFlat = &Gl[0][0];
      const float* gbase = G + (size_t)S * NS;
      for (int idx = tid; idx < (16 * NS) / 4; idx += D) {   // 1024 x b128
        unsigned lds_off = (unsigned)(uintptr_t)(GlFlat + idx * 4);
        unsigned long long gaddr =
            (unsigned long long)(uintptr_t)(gbase + idx * 4);
        asm volatile("global_load_async_to_lds_b128 %0, %1, off"
                     :: "v"(lds_off), "v"(gaddr) : "memory");
      }
      asm volatile("s_wait_asynccnt 0x0" ::: "memory");
    }
    __syncthreads();

    // build coefficient tile (zero-padded to full NS width so WMMA K-pad is exact)
    for (int idx = tid; idx < 16 * NS; idx += D) {
      int rr = idx >> 8, s = idx & (NS - 1);
      int t = S + rr;
      float v = (s < S) ? (ETA * dp[t - 1 - s] * Gl[rr][s]) : 0.f;
      coef[rr][s] = f2bf(v);
    }
    __syncthreads();

    if (c == 0) {
      for (int idx = tid; idx < 16 * D; idx += D) ((bf16*)Pl)[idx] = f2bf(0.f);
    } else {
      const int kmax = (S + 31) & ~31;
      const int tn0 = wave * 2;
      const bf16* Bp0 = HT + (size_t)(tn0 * 16 + r) * NS + hi * 16;
      const bf16* Bp1 = Bp0 + (size_t)16 * NS;
      v8f acc0 = {0.f,0.f,0.f,0.f,0.f,0.f,0.f,0.f};
      v8f acc1 = {0.f,0.f,0.f,0.f,0.f,0.f,0.f,0.f};
      for (int sk = 0; sk < kmax; sk += 32) {
        v16bf a  = cat16(load8(&coef[r][sk + hi * 16]),
                         load8(&coef[r][sk + hi * 16 + 8]));
        v16bf b0 = cat16(load8(Bp0 + sk), load8(Bp0 + sk + 8));
        v16bf b1 = cat16(load8(Bp1 + sk), load8(Bp1 + sk + 8));
        acc0 = __builtin_amdgcn_wmma_f32_16x16x32_bf16(false, a, false, b0,
                                                       (short)0, acc0, false, false);
        acc1 = __builtin_amdgcn_wmma_f32_16x16x32_bf16(false, a, false, b1,
                                                       (short)0, acc1, false, false);
      }
#pragma unroll
      for (int i = 0; i < 8; ++i) {
        Pl[i + hi * 8][tn0 * 16 + r]      = f2bf(acc0[i]);
        Pl[i + hi * 8][tn0 * 16 + 16 + r] = f2bf(acc1[i]);
      }
    }
    __syncthreads();

    // intra-chunk: each thread owns column d; h history stays in registers
    {
      const int d = tid;
      float h[16];
#pragma unroll
      for (int tp = 0; tp < 16; ++tp) {
        const int t = S + tp;
        float v = Z[(size_t)t * D + d] + bf2f(Pl[tp][d]);
        for (int sp = 0; sp < tp; ++sp)
          v += ETA * dp[tp - 1 - sp] * Gl[tp][S + sp] * h[sp];
        v = (act == 1) ? fmaxf(v, 0.f) : 1.f / (1.f + __expf(-v));
        h[tp] = v;
        if (outBf) outBf[(size_t)t * D + d] = f2bf(v);
        if (outF)  outF [(size_t)t * D + d] = v;
        HT[(size_t)d * NS + t] = f2bf(v);
      }
    }
    __threadfence();   // make HT writes visible before next chunk's WMMA reads
    __syncthreads();
  }
}

extern "C" void kernel_launch(void* const* d_in, const int* in_sizes, int n_in,
                              void* d_out, int out_size, void* d_ws, size_t ws_size,
                              hipStream_t stream) {
  const float* X  = (const float*)d_in[0];
  const float* W0 = (const float*)d_in[1];
  const float* b0 = (const float*)d_in[2];
  const float* W1 = (const float*)d_in[3];
  const float* b1 = (const float*)d_in[4];
  const float* W2 = (const float*)d_in[5];
  const float* b2 = (const float*)d_in[6];
  const float* W3 = (const float*)d_in[7];
  const float* b3 = (const float*)d_in[8];

  char* ws = (char*)d_ws;
  size_t off = 0;
  auto alloc = [&](size_t bytes) -> char* {
    char* p = ws + off; off += (bytes + 255) & ~(size_t)255; return p;
  };
  bf16* Xbf   = (bf16*)alloc((size_t)NS * D * 2);
  bf16* W0t   = (bf16*)alloc((size_t)D * D * 2);
  bf16* W1t   = (bf16*)alloc((size_t)D * D * 2);
  bf16* W2t   = (bf16*)alloc((size_t)D * D * 2);
  bf16* W3t   = (bf16*)alloc((size_t)D * D * 2);
  bf16* H0bf  = (bf16*)alloc((size_t)NS * D * 2);
  bf16* H1bf  = (bf16*)alloc((size_t)NS * D * 2);
  bf16* H2bf  = (bf16*)alloc((size_t)NS * D * 2);
  bf16* H2T   = (bf16*)alloc((size_t)D * NS * 2);
  bf16* H3T   = (bf16*)alloc((size_t)D * NS * 2);
  float* Z2   = (float*)alloc((size_t)NS * D * 4);
  float* Z3   = (float*)alloc((size_t)NS * D * 4);
  float* G1   = (float*)alloc((size_t)NS * NS * 4);
  float* G2   = (float*)alloc((size_t)NS * NS * 4);
  (void)ws_size; (void)in_sizes; (void)n_in; (void)out_size;

  // precision staging (one-time, trivial traffic)
  k_cvt<<<(NS * D + 255) / 256, 256, 0, stream>>>(X, Xbf, NS * D);
  k_transpose_cvt<<<(D * D + 255) / 256, 256, 0, stream>>>(W0, W0t);
  k_transpose_cvt<<<(D * D + 255) / 256, 256, 0, stream>>>(W1, W1t);
  k_transpose_cvt<<<(D * D + 255) / 256, 256, 0, stream>>>(W2, W2t);
  k_transpose_cvt<<<(D * D + 255) / 256, 256, 0, stream>>>(W3, W3t);
  k_zero_bf<<<(D * NS + 255) / 256, 256, 0, stream>>>(H2T, D * NS);
  k_zero_bf<<<(D * NS + 255) / 256, 256, 0, stream>>>(H3T, D * NS);

  const int gMain = (NS / 16) * (D / 32) / 8;   // 48 blocks (16x32 per wave)
  const int gGram = (NS / 16) * (NS / 32) / 8;  // 16 blocks

  // H0 = relu(X W0 + b0) ; H1 = relu(H0 W1 + b1)
  k_gemm<<<gMain, 256, 0, stream>>>(Xbf,  W0t, b0, nullptr, H0bf, NS, D, D, 1);
  k_gemm<<<gMain, 256, 0, stream>>>(H0bf, W1t, b1, nullptr, H1bf, NS, D, D, 1);
  // Z2 = H1 W2 + b2 ; G1 = H1 H1^T  (Bt of H1^T is H1 itself)
  k_gemm<<<gMain, 256, 0, stream>>>(H1bf, W2t, b2, Z2, nullptr, NS, D, D, 0);
  k_gemm<<<gGram, 256, 0, stream>>>(H1bf, H1bf, nullptr, G1, nullptr, NS, NS, D, 0);
  // layer-2 Hebbian scan (relu)
  k_scan<<<1, 768, 0, stream>>>(Z2, G1, H2T, H2bf, nullptr, 1);
  // Z3 = H2 W3 + b3 ; G2 = H2 H2^T
  k_gemm<<<gMain, 256, 0, stream>>>(H2bf, W3t, b3, Z3, nullptr, NS, D, D, 0);
  k_gemm<<<gGram, 256, 0, stream>>>(H2bf, H2bf, nullptr, G2, nullptr, NS, NS, D, 0);
  // layer-3 Hebbian scan (sigmoid) -> d_out
  k_scan<<<1, 768, 0, stream>>>(Z3, G2, H3T, nullptr, (float*)d_out, 2);
}
